// cacheLSTM_43112881717459
// MI455X (gfx1250) — matrix operand order repaired and checked by reading, not compiled, not run on hardware
//
#include <hip/hip_runtime.h>
#include <hip/hip_bf16.h>

// Fused single-step LSTM cell for MI455X (gfx1250, wave32).
// Pre-pass (d_ws, needs 80MB):
//   Bcat[n][k] = (k<1024 ? U[k][n] : W[n][k-1024])  bf16 (4096 x 2048)  @ ws+0
//   Acat[b][k] = (k<1024 ? X[b][k] : h0[b][k-1024]) bf16 (16384 x 2048) @ ws+16MB
// Main: gates = Acat @ Bcat^T, K=2048, f32 accum via v_wmma_f32_16x16x32_bf16.
//   A tiles: global_load_async_to_lds_b128 (ASYNCcnt)
//   B tiles: tensor_load_to_lds TDM descriptors w/ LDS padding (TENSORcnt)
//   Double-buffered LDS ping-pong; fused LSTM epilogue.

typedef __bf16 bf16;
typedef __attribute__((ext_vector_type(4)))  __bf16 bf16x4;
typedef __attribute__((ext_vector_type(8)))  __bf16 bf16x8;
typedef __attribute__((ext_vector_type(16))) __bf16 v16bf;
typedef __attribute__((ext_vector_type(8)))  float  v8f;
typedef __attribute__((ext_vector_type(4)))  unsigned int v4u;
typedef __attribute__((ext_vector_type(8)))  int v8i;
typedef __attribute__((ext_vector_type(4)))  int v4i;

#define BATCH 16384
#define DH    1024
#define N4    4096
#define KTOT  2048
#define BM    128      // batch rows per block
#define BN    32       // Dh cols per block (x4 gates)
#define BK    32       // K step
#define LDP   40       // padded LDS row stride in bf16 (80B = 64B data + 16B pad)

__device__ __forceinline__ float fsigmoid(float x) {
    return 1.0f / (1.0f + __expf(-x));
}
__device__ __forceinline__ float ftanh(float x) {
    float e = __expf(2.0f * x);
    return (e - 1.0f) / (e + 1.0f);
}

// CDNA5 async copy: global -> LDS, 16B per lane, tracked by ASYNCcnt.
__device__ __forceinline__ void async_load_b128(unsigned int lds_addr, const void* gaddr) {
    asm volatile("global_load_async_to_lds_b128 %0, %1, off"
                 :: "v"(lds_addr), "v"(gaddr)
                 : "memory");
}
__device__ __forceinline__ void wait_async0() {
    asm volatile("s_wait_asynccnt 0" ::: "memory");
}

// CDNA5 Tensor Data Mover: one 2D tile (BN rows x BK bf16) global->LDS.
// D# packing per CDNA5 ISA:
//   pad_enable=1, pad_interval code 3 (every 16 DWORDs = one 64B row),
//   pad_amount  code 3 (4 DWORDs = 16B) -> LDS row stride 80B = LDP.
__device__ __forceinline__ unsigned int rfl(unsigned int x) {
    return (unsigned int)__builtin_amdgcn_readfirstlane((int)x);
}
__device__ __forceinline__ void tdm_load_tile(const bf16* gsrc, unsigned int lds_addr) {
    unsigned long long ga = (unsigned long long)(uintptr_t)gsrc;
    // Group 0: count=1 | lds_addr | global_addr[56:0] | type=2
    unsigned int g0w0 = 1u;
    unsigned int g0w1 = lds_addr;
    unsigned int g0w2 = (unsigned int)ga;
    unsigned int g0w3 = (unsigned int)((ga >> 32) & 0x1FFFFFFu) | (2u << 30);
    // Group 1: data_size=1(2B) | pad_enable | pad_interval=3 | pad_amount=3
    unsigned int g1w0 = (1u << 16) | (1u << 20) | (3u << 22) | (3u << 25);
    unsigned int g1w1 = ((unsigned int)KTOT & 0xFFFFu) << 16;   // tensor_dim0[15:0]
    unsigned int g1w2 = ((unsigned int)N4  & 0xFFFFu) << 16;    // dim0 hi=0 | tensor_dim1[15:0]
    unsigned int g1w3 = ((unsigned int)BK) << 16;               // dim1 hi=0 | tile_dim0=32
    unsigned int g1w4 = (unsigned int)BN;                       // tile_dim1=32, tile_dim2=0
    unsigned int g1w5 = (unsigned int)KTOT;                     // tensor_dim0_stride lo
    v4u g0 = { rfl(g0w0), rfl(g0w1), rfl(g0w2), rfl(g0w3) };
    v8i g1 = { (int)rfl(g1w0), (int)rfl(g1w1), (int)rfl(g1w2), (int)rfl(g1w3),
               (int)rfl(g1w4), (int)rfl(g1w5), 0, 0 };
    v4i gz = { 0, 0, 0, 0 };
#if __clang_major__ >= 23
    v8i gz8 = { 0, 0, 0, 0, 0, 0, 0, 0 };
    __builtin_amdgcn_tensor_load_to_lds(g0, g1, gz, gz, gz8, 0);
#else
    __builtin_amdgcn_tensor_load_to_lds(g0, g1, gz, gz, 0);
#endif
}

// ---------- Pre-pass 1: W (4096,1024) f32 -> Bcat[n][1024+k] bf16 ----------
__global__ __launch_bounds__(256)
void prep_w(const float* __restrict__ W, bf16* __restrict__ Bcat) {
    int idx = blockIdx.x * 256 + threadIdx.x;
    int n  = idx >> 8;
    int kc = (idx & 255) << 2;
    const float4 v = *(const float4*)&W[(size_t)n * DH + kc];
    bf16x4 pk = { (bf16)v.x, (bf16)v.y, (bf16)v.z, (bf16)v.w };
    *(bf16x4*)&Bcat[(size_t)n * KTOT + DH + kc] = pk;
}

// ---------- Pre-pass 2: U (1024,4096) f32 -> Bcat[n][k] bf16 (LDS transpose) ----------
__global__ __launch_bounds__(256)
void prep_u(const float* __restrict__ U, bf16* __restrict__ Bcat) {
    __shared__ float t[32][33];
    const int kb = blockIdx.x * 32;
    const int nb = blockIdx.y * 32;
    const int tid = threadIdx.x;
    {
        int k  = tid >> 3;
        int nc = (tid & 7) << 2;
        const float4 v = *(const float4*)&U[(size_t)(kb + k) * N4 + nb + nc];
        t[k][nc + 0] = v.x; t[k][nc + 1] = v.y; t[k][nc + 2] = v.z; t[k][nc + 3] = v.w;
    }
    __syncthreads();
    {
        int n  = tid >> 3;
        int kc = (tid & 7) << 2;
        bf16x4 pk = { (bf16)t[kc + 0][n], (bf16)t[kc + 1][n],
                      (bf16)t[kc + 2][n], (bf16)t[kc + 3][n] };
        *(bf16x4*)&Bcat[(size_t)(nb + n) * KTOT + kb + kc] = pk;
    }
}

// ---------- Pre-pass 3: X|h0 f32 -> Acat[b][k] bf16 (16384 x 2048) ----------
__global__ __launch_bounds__(256)
void prep_a(const float* __restrict__ X, const float* __restrict__ H0,
            bf16* __restrict__ Acat) {
    size_t idx = (size_t)blockIdx.x * 256 + threadIdx.x;   // float4 chunks
    int b  = (int)(idx >> 9);                              // 512 chunks per row
    int kc = ((int)idx & 511) << 2;                        // 0..2044
    const float* src = (kc < DH) ? &X[(size_t)b * DH + kc]
                                 : &H0[(size_t)b * DH + (kc - DH)];
    const float4 v = *(const float4*)src;
    bf16x4 pk = { (bf16)v.x, (bf16)v.y, (bf16)v.z, (bf16)v.w };
    *(bf16x4*)&Acat[(size_t)b * KTOT + kc] = pk;
}

// ---------- Main fused GEMM + LSTM ----------
__global__ __launch_bounds__(256, 2)
void lstm_wmma_fused(const bf16* __restrict__ Acat,
                     const bf16* __restrict__ Bcat,
                     const float* __restrict__ C0,
                     float* __restrict__ out) {
    __shared__ bf16 As[2][BM][LDP];       // ping-pong A tiles
    __shared__ bf16 Bs[2][4][BN][LDP];    // ping-pong B tiles (TDM-padded rows)

    const int tid  = threadIdx.x;
    const int lane = tid & 31;
    const int wave = tid >> 5;            // 0..7
    const int mrow = wave >> 1;           // 0..3 -> two 16-row subtiles each
    const int ni   = wave & 1;            // 0..1 -> 16-col subtile
    const int lm   = lane & 15;
    const int half = lane >> 4;

    const int rowBase = blockIdx.y * BM;
    const int colBase = blockIdx.x * BN;

    v8f acc[2][4];
    #pragma unroll
    for (int m2 = 0; m2 < 2; ++m2)
        #pragma unroll
        for (int g = 0; g < 4; ++g)
            acc[m2][g] = (v8f){0.f,0.f,0.f,0.f,0.f,0.f,0.f,0.f};

    auto stage = [&](int kk, int q) {
        // A tile (128 x 32 bf16): pure async copies, 16B per lane (ASYNCcnt).
        #pragma unroll
        for (int i = 0; i < 2; ++i) {
            int c   = tid + i * 256;              // 0..511
            int row = c >> 2;                     // 0..127
            int kc  = (c & 3) << 3;               // 0,8,16,24 bf16
            unsigned int lds = (unsigned int)(uintptr_t)&As[q][row][kc];
            async_load_b128(lds, Acat + (size_t)(rowBase + row) * KTOT + kk + kc);
        }
        // B tiles: one TDM descriptor per gate, issued by waves 0..3 (TENSORcnt).
        if (wave < 4) {
            const bf16* gsrc = Bcat + (size_t)(wave * DH + colBase) * KTOT + kk;
            tdm_load_tile(gsrc, (unsigned int)(uintptr_t)&Bs[q][wave][0][0]);
        }
    };

    // Prologue: fill buffer 0
    stage(0, 0);
    wait_async0();
    if (wave < 4) __builtin_amdgcn_s_wait_tensorcnt(0);
    __syncthreads();

    int p = 0;
    for (int kk = 0; kk < KTOT; kk += BK) {
        if (kk + BK < KTOT) stage(kk + BK, p ^ 1);

        // A fragments (16x32): ISA layout K = 16*(e/8) + 8*half + e%8
        v16bf af[2];
        #pragma unroll
        for (int m2 = 0; m2 < 2; ++m2) {
            const bf16* arow = &As[p][(mrow * 2 + m2) * 16 + lm][0];
            bf16x8 alo = *(const bf16x8*)(arow + 8 * half);
            bf16x8 ahi = *(const bf16x8*)(arow + 16 + 8 * half);
            af[m2] = __builtin_shufflevector(alo, ahi,
                0,1,2,3,4,5,6,7,8,9,10,11,12,13,14,15);
        }
        // Per gate: one B fragment (32x16: K = 16*(e/8) + 8*half + e%8), 2 WMMAs
        #pragma unroll
        for (int g = 0; g < 4; ++g) {
            const bf16* brow = &Bs[p][g][ni * 16 + lm][8 * half];
            bf16x8 blo = *(const bf16x8*)(brow);
            bf16x8 bhi = *(const bf16x8*)(brow + 16);
            v16bf bfrag = __builtin_shufflevector(blo, bhi,
                0,1,2,3,4,5,6,7,8,9,10,11,12,13,14,15);
            acc[0][g] = __builtin_amdgcn_wmma_f32_16x16x32_bf16(
                false, af[0], false, bfrag, (short)0, acc[0][g], false, false);
            acc[1][g] = __builtin_amdgcn_wmma_f32_16x16x32_bf16(
                false, af[1], false, bfrag, (short)0, acc[1][g], false, false);
        }

        wait_async0();                                     // A copies (next buf) landed
        if (wave < 4) __builtin_amdgcn_s_wait_tensorcnt(0); // B TDM (next buf) landed
        __syncthreads();
        p ^= 1;
    }

    // Fused LSTM epilogue. C/D layout: N = lm, VGPR r -> M = r + 8*half
    const int gn = colBase + ni * 16 + lm;
    #pragma unroll
    for (int m2 = 0; m2 < 2; ++m2) {
        #pragma unroll
        for (int r = 0; r < 8; ++r) {
            const int gm = rowBase + (mrow * 2 + m2) * 16 + r + 8 * half;
            const size_t off = (size_t)gm * DH + gn;
            float iv = acc[m2][0][r];
            float fv = acc[m2][1][r];
            float gv = acc[m2][2][r];
            float ov = acc[m2][3][r];
            float c0v = C0[off];
            float cv = fsigmoid(fv) * c0v + fsigmoid(iv) * ftanh(gv);
            float hv = fsigmoid(ov) * ftanh(cv);
            out[off] = hv;                           // h
            out[(size_t)BATCH * DH + off] = cv;      // c
        }
    }
}

extern "C" void kernel_launch(void* const* d_in, const int* in_sizes, int n_in,
                              void* d_out, int out_size, void* d_ws, size_t ws_size,
                              hipStream_t stream) {
    (void)in_sizes; (void)n_in; (void)out_size; (void)ws_size;  // ws must be >= 80MB
    const float* X  = (const float*)d_in[0];
    const float* H0 = (const float*)d_in[1];
    const float* C0 = (const float*)d_in[2];
    const float* U  = (const float*)d_in[3];
    const float* W  = (const float*)d_in[4];
    float* out = (float*)d_out;
    bf16* Bcat = (bf16*)d_ws;                                   // 16 MB
    bf16* Acat = (bf16*)((char*)d_ws + (size_t)N4 * KTOT * 2);  // 64 MB

    // Pre-pass: bf16 operand matrices, K-contiguous.
    prep_w<<<dim3(N4 * (DH / 4) / 256), dim3(256), 0, stream>>>(W, Bcat);
    prep_u<<<dim3(DH / 32, N4 / 32), dim3(256), 0, stream>>>(U, Bcat);
    prep_a<<<dim3((size_t)BATCH * (KTOT / 4) / 256), dim3(256), 0, stream>>>(X, H0, Acat);

    // Main fused GEMM + LSTM.
    dim3 grid(DH / BN, BATCH / BM);   // (32, 128)
    lstm_wmma_fused<<<grid, dim3(256), 0, stream>>>(Acat, Bcat, C0, out);
}
